// BEVFormerLite_63136019251322
// MI455X (gfx1250) — matrix operand order repaired and checked by reading, not compiled, not run on hardware
//
#include <hip/hip_runtime.h>
#include <hip/hip_bf16.h>
#include <stdint.h>

#define BATCH 4
#define NCAM  6
#define CH    256
#define FH    29
#define FW    50
#define FHW   (FH * FW)      // 1450
#define BEVW  200
#define BEVP  (BEVW * BEVW)  // 40000

typedef __attribute__((ext_vector_type(16))) __bf16         v16bf;
typedef __attribute__((ext_vector_type(8)))  float          v8f;
typedef __attribute__((ext_vector_type(16))) unsigned short v16u;
typedef __attribute__((ext_vector_type(8)))  unsigned short v8u;

__device__ __forceinline__ unsigned short f2bf(float f) {
    union { float f; unsigned u; } v; v.f = f;
    unsigned u = v.u;
    unsigned r = u + 0x7FFFu + ((u >> 16) & 1u);       // round-to-nearest-even
    if ((u & 0x7F800000u) == 0x7F800000u) r = u;       // keep inf/nan
    return (unsigned short)(r >> 16);
}
__device__ __forceinline__ float bf2f(unsigned short h) {
    union { unsigned u; float f; } v; v.u = ((unsigned)h) << 16;
    return v.f;
}

// ---------------------------------------------------------------------------
// 1) Per-camera projection setup: M = K * inv(R), v = K * t', depth row.
// ---------------------------------------------------------------------------
__global__ void k_prep_cams(const float* __restrict__ K, const float* __restrict__ E,
                            float* __restrict__ camP) {
    int i = threadIdx.x;
    if (i >= BATCH * NCAM) return;
    const float* e = E + i * 16;
    float a00=e[0], a01=e[1], a02=e[2],  t0=e[3];
    float a10=e[4], a11=e[5], a12=e[6],  t1=e[7];
    float a20=e[8], a21=e[9], a22=e[10], t2=e[11];
    float c00 = a11*a22 - a12*a21;
    float c01 = a02*a21 - a01*a22;
    float c02 = a01*a12 - a02*a11;
    float c10 = a12*a20 - a10*a22;
    float c11 = a00*a22 - a02*a20;
    float c12 = a02*a10 - a00*a12;
    float c20 = a10*a21 - a11*a20;
    float c21 = a01*a20 - a00*a21;
    float c22 = a00*a11 - a01*a10;
    float det = a00*c00 + a01*c10 + a02*c20;
    float id  = 1.0f / det;
    float i00=c00*id, i01=c01*id, i02=c02*id;
    float i10=c10*id, i11=c11*id, i12=c12*id;
    float i20=c20*id, i21=c21*id, i22=c22*id;
    float tp0 = -(i00*t0 + i01*t1 + i02*t2);
    float tp1 = -(i10*t0 + i11*t1 + i12*t2);
    float tp2 = -(i20*t0 + i21*t1 + i22*t2);
    const float* k = K + i * 9;
    float m00 = k[0]*i00 + k[1]*i10 + k[2]*i20;
    float m01 = k[0]*i01 + k[1]*i11 + k[2]*i21;
    float m10 = k[3]*i00 + k[4]*i10 + k[5]*i20;
    float m11 = k[3]*i01 + k[4]*i11 + k[5]*i21;
    float m20 = k[6]*i00 + k[7]*i10 + k[8]*i20;
    float m21 = k[6]*i01 + k[7]*i11 + k[8]*i21;
    float v0  = k[0]*tp0 + k[1]*tp1 + k[2]*tp2;
    float v1  = k[3]*tp0 + k[4]*tp1 + k[5]*tp2;
    float v2  = k[6]*tp0 + k[7]*tp1 + k[8]*tp2;
    float* o = camP + i * 12;
    o[0]=m00; o[1]=m01; o[2]=m10; o[3]=m11; o[4]=m20; o[5]=m21;
    o[6]=v0;  o[7]=v1;  o[8]=v2;  o[9]=i20; o[10]=i21; o[11]=tp2;
}

// ---------------------------------------------------------------------------
// 2) Fold conv bias + BN into per-channel affine.
// ---------------------------------------------------------------------------
__global__ void k_prep_affine(const float* __restrict__ cb, const float* __restrict__ g,
                              const float* __restrict__ bt, const float* __restrict__ mn,
                              const float* __restrict__ vr,
                              float* __restrict__ scale, float* __restrict__ shift) {
    int o = threadIdx.x;
    float s = g[o] * rsqrtf(vr[o] + 1e-5f);
    scale[o] = s;
    shift[o] = bt[o] + (cb[o] - mn[o]) * s;
}

// ---------------------------------------------------------------------------
// 3) conv_w fp32 -> bf16 (row-major [o][c], K-contiguous for the GEMM A tile).
// ---------------------------------------------------------------------------
__global__ void k_w2bf(const float* __restrict__ w, unsigned short* __restrict__ wbf) {
    int i = blockIdx.x * 256 + threadIdx.x;
    wbf[i] = f2bf(w[i]);
}

// ---------------------------------------------------------------------------
// 4) feats (b,n,c,h,w) fp32 -> featsT (b,n,h,w,c) bf16, tiled 32x32 via LDS.
// ---------------------------------------------------------------------------
__global__ void k_transpose(const float* __restrict__ feats, unsigned short* __restrict__ ft) {
    __shared__ float tile[32][33];
    int bn  = blockIdx.z;
    int hw0 = blockIdx.x * 32;
    int c0  = blockIdx.y * 32;
    int tx  = threadIdx.x & 31;
    int ty  = threadIdx.x >> 5;  // 0..7
    const float* src = feats + (size_t)bn * CH * FHW;
    #pragma unroll
    for (int i = 0; i < 4; ++i) {
        int c  = c0 + ty + i * 8;
        int hw = hw0 + tx;
        float v = 0.f;
        if (hw < FHW) v = src[(size_t)c * FHW + hw];
        tile[ty + i * 8][tx] = v;
    }
    __syncthreads();
    unsigned short* dst = ft + (size_t)bn * FHW * CH;
    #pragma unroll
    for (int i = 0; i < 4; ++i) {
        int hw = hw0 + ty + i * 8;
        int c  = c0 + tx;
        if (hw < FHW) dst[(size_t)hw * CH + c] = f2bf(tile[tx][ty + i * 8]);
    }
}

// ---------------------------------------------------------------------------
// 5) Project + bilinear-sample + camera-average -> bev bf16 (b, p, c).
// ---------------------------------------------------------------------------
__global__ void __launch_bounds__(256) k_sample(const unsigned short* __restrict__ ft,
                                                const float* __restrict__ camP,
                                                unsigned short* __restrict__ bev) {
    int p = blockIdx.x;
    int b = blockIdx.y;
    __shared__ float sw[NCAM][4];
    __shared__ int   soff[NCAM][4];
    __shared__ float sval[NCAM];
    int tid = threadIdx.x;
    if (tid < NCAM) {
        int n = tid;
        const float* cp = camP + (b * NCAM + n) * 12;
        int iy = p / BEVW, ix = p - iy * BEVW;
        float x = -49.75f + 0.5f * (float)ix;
        float y = -49.75f + 0.5f * (float)iy;
        float pix   = cp[0]*x + cp[1]*y + cp[6];
        float piy   = cp[2]*x + cp[3]*y + cp[7];
        float piz   = cp[4]*x + cp[5]*y + cp[8];
        float depth = cp[9]*x + cp[10]*y + cp[11];
        float u  = pix / (piz + 1e-6f);
        float v  = piy / (piz + 1e-6f);
        float xs = u * ((float)FW / 1600.0f);
        float ys = v * ((float)FH / 928.0f);
        bool valid = (depth > 0.1f) && (xs >= 0.f) && (xs <= (float)(FW - 1)) &&
                     (ys >= 0.f) && (ys <= (float)(FH - 1));
        float w00=0.f, w01=0.f, w10=0.f, w11=0.f;
        int   o00=0,   o01=0,   o10=0,   o11=0;
        if (valid) {
            float x0f = floorf(xs), y0f = floorf(ys);
            float wx = xs - x0f, wy = ys - y0f;
            int x0 = (int)x0f, y0 = (int)y0f;
            int x1 = x0 + 1,   y1 = y0 + 1;
            bool okx0 = (x0 >= 0 && x0 < FW), okx1 = (x1 >= 0 && x1 < FW);
            bool oky0 = (y0 >= 0 && y0 < FH), oky1 = (y1 >= 0 && y1 < FH);
            int cx0 = min(max(x0, 0), FW - 1), cx1 = min(max(x1, 0), FW - 1);
            int cy0 = min(max(y0, 0), FH - 1), cy1 = min(max(y1, 0), FH - 1);
            int base = (b * NCAM + n) * FHW;
            o00 = (base + cy0 * FW + cx0) * CH;
            o01 = (base + cy0 * FW + cx1) * CH;
            o10 = (base + cy1 * FW + cx0) * CH;
            o11 = (base + cy1 * FW + cx1) * CH;
            w00 = (okx0 && oky0) ? (1.f - wx) * (1.f - wy) : 0.f;
            w01 = (okx1 && oky0) ? wx * (1.f - wy)         : 0.f;
            w10 = (okx0 && oky1) ? (1.f - wx) * wy         : 0.f;
            w11 = (okx1 && oky1) ? wx * wy                 : 0.f;
        }
        sw[n][0]=w00; sw[n][1]=w01; sw[n][2]=w10; sw[n][3]=w11;
        soff[n][0]=o00; soff[n][1]=o01; soff[n][2]=o10; soff[n][3]=o11;
        sval[n] = valid ? 1.f : 0.f;
    }
    __syncthreads();
    float vcnt = sval[0]+sval[1]+sval[2]+sval[3]+sval[4]+sval[5];
    float inv  = 1.0f / (vcnt + 1e-6f);
    int c = tid;
    float acc = 0.f;
    #pragma unroll
    for (int n = 0; n < NCAM; ++n) {
        acc += sw[n][0] * bf2f(ft[soff[n][0] + c]);
        acc += sw[n][1] * bf2f(ft[soff[n][1] + c]);
        acc += sw[n][2] * bf2f(ft[soff[n][2] + c]);
        acc += sw[n][3] * bf2f(ft[soff[n][3] + c]);
    }
    bev[((size_t)b * BEVP + p) * CH + c] = f2bf(acc * inv);
}

// ---------------------------------------------------------------------------
// 6) GEMM + affine + ReLU, bf16 WMMA 16x16x32, f32 accumulate.
//    Block = 8 waves, tile 128(M)x64(N). Double-buffered LDS filled with
//    CDNA5 async global->LDS copies (ASYNCcnt), no VGPR round-trip.
// ---------------------------------------------------------------------------
#define GEMM_LDA   48                      // padded K-stride (elements)
#define GEMM_ABUF  (128 * GEMM_LDA)        // elements
#define GEMM_BBUF  (64 * GEMM_LDA)
#define GEMM_BUFSZ (GEMM_ABUF + GEMM_BBUF) // 9216 elements = 18432 B
#define GEMM_LDS_BYTES (2 * GEMM_BUFSZ * 2)

__global__ void __launch_bounds__(256) k_gemm(const unsigned short* __restrict__ Wbf,
                                              const unsigned short* __restrict__ bev,
                                              const float* __restrict__ scale,
                                              const float* __restrict__ shift,
                                              float* __restrict__ out) {
    constexpr int LDA = GEMM_LDA;
    extern __shared__ __align__(16) unsigned short smem[];  // 2 x (A 128x32 | B 64x32), padded

    int tid   = threadIdx.x;
    int b     = blockIdx.z;
    int mBase = blockIdx.y * 128;
    int pBase = blockIdx.x * 64;
    int lane  = tid & 31;
    int wid   = tid >> 5;
    int waveM = (wid & 3) * 32;
    int waveN = (wid >> 2) * 32;
    int nrow  = lane & 15;
    int khalf = lane >> 4;

    v8f acc[2][2] = {};

    const unsigned short* Aglob = Wbf + (size_t)mBase * CH;
    const unsigned short* Bglob = bev + ((size_t)b * BEVP + pBase) * CH;

    int lr = tid >> 2;        // 0..63 : row within pass
    int lc = (tid & 3) * 8;   // 0,8,16,24 : 8 bf16 (16 B) per thread

    // Issue async global->LDS b128 copies for K-chunk `ks` into buffer `buf`.
    // LDS offset = low 32 bits of the flat LDS-aperture address (wave-relative).
    auto issue = [&](int ks, int buf) {
        unsigned short* a   = smem + buf * GEMM_BUFSZ;
        unsigned short* bsh = a + GEMM_ABUF;
        #pragma unroll
        for (int pass = 0; pass < 2; ++pass) {      // A: 128 rows x 32 k
            int r = pass * 64 + lr;
            unsigned           lo = (unsigned)(uintptr_t)(a + r * LDA + lc);
            unsigned long long ga = (unsigned long long)(uintptr_t)(Aglob + (size_t)r * CH + ks + lc);
            asm volatile("global_load_async_to_lds_b128 %0, %1, off"
                         :: "v"(lo), "v"(ga) : "memory");
        }
        {                                           // B: 64 rows x 32 k
            int r = lr;
            unsigned           lo = (unsigned)(uintptr_t)(bsh + r * LDA + lc);
            unsigned long long ga = (unsigned long long)(uintptr_t)(Bglob + (size_t)r * CH + ks + lc);
            asm volatile("global_load_async_to_lds_b128 %0, %1, off"
                         :: "v"(lo), "v"(ga) : "memory");
        }
    };

    issue(0, 0);
    asm volatile("s_wait_asynccnt 0x0" ::: "memory");
    __syncthreads();

    for (int i = 0; i < CH / 32; ++i) {
        int cur = i & 1;
        if (i + 1 < CH / 32) {
            issue((i + 1) * 32, cur ^ 1);           // prefetch next chunk via async engine
            __builtin_prefetch((const void*)(Aglob + (size_t)lr * CH + (i + 1) * 32 + lc), 0, 1);
        }

        const unsigned short* As = smem + cur * GEMM_BUFSZ;
        const unsigned short* Bs = As + GEMM_ABUF;

        // Build fragments in contiguous k order (k = 16*khalf + s), same for A and B.
        v16bf afrag[2], bfrag[2];
        #pragma unroll
        for (int mi = 0; mi < 2; ++mi) {
            int row = waveM + mi * 16 + nrow;  // A: lane -> M row
            v8u loh = *(const v8u*)(As + row * LDA + khalf * 16);       // K 0-7 / 16-23
            v8u hih = *(const v8u*)(As + row * LDA + khalf * 16 + 8);   // K 8-15 / 24-31
            v16u t;
            #pragma unroll
            for (int e = 0; e < 8; ++e) { t[e] = loh[e]; t[e + 8] = hih[e]; }
            afrag[mi] = __builtin_bit_cast(v16bf, t);
        }
        #pragma unroll
        for (int ni = 0; ni < 2; ++ni) {
            int row = waveN + ni * 16 + nrow;  // B: lane -> N column; contiguous 16 K
            v8u loh = *(const v8u*)(Bs + row * LDA + khalf * 16);
            v8u hih = *(const v8u*)(Bs + row * LDA + khalf * 16 + 8);
            v16u t;
            #pragma unroll
            for (int e = 0; e < 8; ++e) { t[e] = loh[e]; t[e + 8] = hih[e]; }
            bfrag[ni] = __builtin_bit_cast(v16bf, t);
        }
        #pragma unroll
        for (int mi = 0; mi < 2; ++mi)
            #pragma unroll
            for (int ni = 0; ni < 2; ++ni)
                acc[mi][ni] = __builtin_amdgcn_wmma_f32_16x16x32_bf16(
                    false, afrag[mi], false, bfrag[ni],
                    (short)0, acc[mi][ni], false, false);

        if (i + 1 < CH / 32)
            asm volatile("s_wait_asynccnt 0x0" ::: "memory");  // next buffer filled
        __syncthreads();                                       // all reads of cur done
    }

    // Epilogue: C/D layout -> (o, p), fused affine + ReLU.
    #pragma unroll
    for (int mi = 0; mi < 2; ++mi) {
        #pragma unroll
        for (int ni = 0; ni < 2; ++ni) {
            int pcol = pBase + waveN + ni * 16 + nrow;
            #pragma unroll
            for (int j = 0; j < 8; ++j) {
                int o = mBase + waveM + mi * 16 + j + 8 * khalf;
                float v = acc[mi][ni][j] * scale[o] + shift[o];
                v = v > 0.f ? v : 0.f;
                out[((size_t)b * CH + o) * BEVP + pcol] = v;
            }
        }
    }
}

// ---------------------------------------------------------------------------
// Launcher
// ---------------------------------------------------------------------------
extern "C" void kernel_launch(void* const* d_in, const int* in_sizes, int n_in,
                              void* d_out, int out_size, void* d_ws, size_t ws_size,
                              hipStream_t stream) {
    const float* feats = (const float*)d_in[0];
    const float* intr  = (const float*)d_in[1];
    const float* extr  = (const float*)d_in[2];
    const float* convw = (const float*)d_in[3];
    const float* convb = (const float*)d_in[4];
    const float* gamma = (const float*)d_in[5];
    const float* beta  = (const float*)d_in[6];
    const float* mean  = (const float*)d_in[7];
    const float* var   = (const float*)d_in[8];
    float* out = (float*)d_out;

    char* ws = (char*)d_ws;
    float*          camP   = (float*)(ws + 0);                       //  1152 B
    float*          scale  = (float*)(ws + 4096);                    //  1024 B
    float*          shift  = (float*)(ws + 4096 + 1024);             //  1024 B
    unsigned short* Wbf    = (unsigned short*)(ws + 8192);           // 128 KB
    unsigned short* featsT = (unsigned short*)(ws + 139264);         // ~17.8 MB
    unsigned short* bev    = (unsigned short*)(ws + 17956864ull);    // ~81.9 MB

    k_prep_cams<<<1, 32, 0, stream>>>(intr, extr, camP);
    k_prep_affine<<<1, 256, 0, stream>>>(convb, gamma, beta, mean, var, scale, shift);
    k_w2bf<<<(CH * CH) / 256, 256, 0, stream>>>(convw, Wbf);

    dim3 tgrid((FHW + 31) / 32, CH / 32, BATCH * NCAM);
    k_transpose<<<tgrid, 256, 0, stream>>>(feats, featsT);

    dim3 sgrid(BEVP, BATCH);
    k_sample<<<sgrid, 256, 0, stream>>>(featsT, camP, bev);

    dim3 ggrid(BEVP / 64, 2, BATCH);   // 625 x 2 x 4
    k_gemm<<<ggrid, 256, GEMM_LDS_BYTES, stream>>>(Wbf, bev, scale, shift, out);
}
